// SpatialTemporalTensorAttention_85109071937558
// MI455X (gfx1250) — matrix operand order repaired and checked by reading, not compiled, not run on hardware
//
#include <hip/hip_runtime.h>
#include <hip/hip_bf16.h>

#define DEV __device__ __forceinline__

typedef _Float16 v16h __attribute__((ext_vector_type(16)));
typedef _Float16 v8h  __attribute__((ext_vector_type(8)));
typedef float    v8f  __attribute__((ext_vector_type(8)));

namespace stta {

constexpr int Bn = 8, T = 128, O = 64, D = 256, H = 8, HD = 32;
constexpr int M_TOTAL = Bn * T * O;                // 65536 tokens
constexpr float QSCALE = 0.17677669529663687f;     // 1/sqrt(HD)

DEV v16h cat8(v8h lo, v8h hi) {
  return __builtin_shufflevector(lo, hi, 0,1,2,3,4,5,6,7,8,9,10,11,12,13,14,15);
}

DEV v8f zero8() { v8f z = {0.f,0.f,0.f,0.f,0.f,0.f,0.f,0.f}; return z; }

DEV v8f wmma(v16h a, v16h b, v8f c) {
  // D = A(16x32 f16) * B(32x16 f16) + C(16x16 f32)
  return __builtin_amdgcn_wmma_f32_16x16x32_f16(false, a, false, b, (short)0, c,
                                                false, false);
}

// A fragment (16x32 f16) from a row-major f16 tile; `ld` = row stride in halves.
// Lane holds row (lane&15); K halves at base + {0..7} and base + {16..23},
// base = (lane>=16 ? 8 : 0)  [ISA 7.12.2, 16-bit A-matrix layout]
DEV v16h load_a_f16(const _Float16* tile, int ld, int l16, int lgrp) {
  const _Float16* p = tile + (size_t)l16 * ld + lgrp * 8;
  v8h lo = *(const v8h*)p;
  v8h hi = *(const v8h*)(p + 16);
  return cat8(lo, hi);
}

// B fragment (32x16 f16) from a source where row = N index, columns = K
// (i.e. a K-contiguous "transposed" operand such as row-major W, K-matrix,
// or an explicitly transposed V/X tile). Lane holds col (lane&15); 16
// K halves at base + {0..7} and base + {16..23}, base = (lane>=16 ? 8 : 0)
DEV v16h load_b_f16(const _Float16* tile, int ld, int l16, int lgrp) {
  const _Float16* p = tile + (size_t)l16 * ld + lgrp * 8;
  v8h lo = *(const v8h*)p;
  v8h hi = *(const v8h*)(p + 16);
  return cat8(lo, hi);
}

// ---------------------------------------------------------------- weights→f16
__global__ void wconv_kernel(const float* Wt, const float* Ws, const float* Wo,
                             _Float16* wt_h, _Float16* ws_h, _Float16* wo_h) {
  int i = blockIdx.x * 256 + threadIdx.x;
  const int NT = 3 * D * D, NS = 2 * D * D, NO = D * D;
  if (i < NT)                wt_h[i] = (_Float16)Wt[i];
  else if (i < NT + NS)      ws_h[i - NT] = (_Float16)Ws[i - NT];
  else if (i < NT + NS + NO) wo_h[i - NT - NS] = (_Float16)Wo[i - NT - NS];
}

// ------------------------------------------------- fused 5-way projection GEMM
// grid = (M_TOTAL/64, 5); block = 512 (16 waves). Block: 64 rows x 256 cols.
// Wave w: row-tile (w>>2), col-tiles (w&3)*4 .. +3. Only 4 accumulators/wave
// (~90 VGPRs) so the allocator can keep all staged B-frags live and
// software-pipeline loads across the fully unrolled K loop.
__global__ void __launch_bounds__(512)
proj_kernel(const float* q, const float* k, const float* v,
            const _Float16* wt_h, const _Float16* ws_h,
            const float* bt, const float* bs,
            _Float16* qt, _Float16* kt, _Float16* vt,
            _Float16* qs, _Float16* ks) {
  const int chunk = blockIdx.y;
  const float* X; const _Float16* W; const float* bias;
  float scale; _Float16* dst; bool slay;
  switch (chunk) {
    case 0:  X=q; W=wt_h;         bias=bt;       scale=QSCALE; dst=qt; slay=false; break;
    case 1:  X=k; W=wt_h + D*D;   bias=bt + D;   scale=1.f;    dst=kt; slay=false; break;
    case 2:  X=v; W=wt_h + 2*D*D; bias=bt + 2*D; scale=1.f;    dst=vt; slay=false; break;
    case 3:  X=q; W=ws_h;         bias=bs;       scale=QSCALE; dst=qs; slay=true;  break;
    default: X=k; W=ws_h + D*D;   bias=bs + D;   scale=1.f;    dst=ks; slay=true;  break;
  }
  const int tid = threadIdx.x, wave = tid >> 5, lane = tid & 31;
  const int lgrp = lane >> 4, l16 = lane & 15;
  const int m0  = blockIdx.x * 64 + (wave >> 2) * 16;
  const int ct0 = (wave & 3) * 4;

  const float* arow = X + (size_t)(m0 + l16) * D + lgrp * 8;
  const _Float16* wrow = W + ((size_t)(ct0 * 16 + l16)) * D + lgrp * 8;

  v8f acc[4];
#pragma unroll
  for (int j = 0; j < 4; ++j) acc[j] = zero8();

#pragma unroll
  for (int kb = 0; kb < D; kb += 32) {
    // prefetch next k-step of the fp32 activation stream
    if (kb + 32 < D) __builtin_prefetch(arow + kb + 32, 0, 3);
    // A frag: convert fp32 rows on the fly (two 32B loads per lane)
    v8f a0 = *(const v8f*)(arow + kb);
    v8f a1 = *(const v8f*)(arow + kb + 16);
    v16h afrag = cat8(__builtin_convertvector(a0, v8h),
                      __builtin_convertvector(a1, v8h));
    // load phase: all 4 B fragments (8x b128 in flight)
    v16h bf[4];
#pragma unroll
    for (int j = 0; j < 4; ++j)
      bf[j] = load_b_f16(wrow + (size_t)j * 16 * D + kb, 0, 0, 0);
    // math phase
#pragma unroll
    for (int j = 0; j < 4; ++j) acc[j] = wmma(afrag, bf[j], acc[j]);
  }

  // Epilogue: +bias, *scale, f16 store into attention-friendly layout
#pragma unroll
  for (int j = 0; j < 4; ++j) {
    const int n = (ct0 + j) * 16 + l16;      // 0..255 within chunk
    const float bv = bias[n];
    const int h = n >> 5, d = n & (HD - 1);
#pragma unroll
    for (int r = 0; r < 8; ++r) {
      const int m = m0 + lgrp * 8 + r;       // D-frag: lanes16-31 → M=8+r
      const int o  = m & (O - 1);
      const int bt2 = m >> 6;
      const int t  = bt2 & (T - 1);
      const int b  = bt2 >> 7;
      const float val = (acc[j][r] + bv) * scale;
      size_t off;
      if (!slay) off = ((((size_t)b * O + o) * H + h) * T + t) * HD + d; // (B,O,H,T,HD)
      else       off = ((((size_t)b * T + t) * H + h) * O + o) * HD + d; // (B,T,H,O,HD)
      dst[off] = (_Float16)val;
    }
  }
}

// -------------------------------------------------- temporal attention (TxT)
// grid = B*O*H = 4096; block = 256 (8 waves). Wave w owns S row-tile w (16 t's).
__global__ void __launch_bounds__(256)
temporal_kernel(const _Float16* qt, const _Float16* kt, const _Float16* vt,
                _Float16* tmp) {
  const int idx = blockIdx.x;
  const int h = idx & (H - 1), bo = idx >> 3, o = bo & (O - 1), b = bo >> 6;
  const size_t base = (((size_t)b * O + o) * H + h) * (size_t)(T * HD);
  const _Float16* Q = qt + base;   // (t, d) row-major, ld = 32
  const _Float16* K = kt + base;
  const _Float16* V = vt + base;

  __shared__ alignas(16) _Float16 Vt[HD * T];      // transposed V: [d][s]
  __shared__ alignas(16) _Float16 P[8][16 * T];    // per-wave probs (16 x 128)

  const int tid = threadIdx.x, wave = tid >> 5, lane = tid & 31;
  const int lgrp = lane >> 4, l16 = lane & 15;

  { // stage V transposed into LDS (each thread: one half-row of V)
    const int s = tid >> 1, d0 = (tid & 1) * 16;
    v8h x0 = *(const v8h*)(V + s * HD + d0);
    v8h x1 = *(const v8h*)(V + s * HD + d0 + 8);
#pragma unroll
    for (int i = 0; i < 8; ++i) {
      Vt[(d0 + i) * T + s]     = x0[i];
      Vt[(d0 + 8 + i) * T + s] = x1[i];
    }
  }
  __syncthreads();

  // S = Q·Kᵀ : one WMMA per 16x16 tile (K dim = HD = 32 exactly)
  const v16h qa = load_a_f16(Q + (size_t)wave * 16 * HD, HD, l16, lgrp);
  v16h kf[8];
#pragma unroll
  for (int j = 0; j < 8; ++j)
    kf[j] = load_b_f16(K + (size_t)j * 16 * HD, HD, l16, lgrp);
  v8f s_acc[8];
#pragma unroll
  for (int j = 0; j < 8; ++j) {
    s_acc[j] = zero8();
    s_acc[j] = wmma(qa, kf[j], s_acc[j]);
  }

  // softmax over full rows (8 tiles x 16 lanes of the half-wave)
#pragma unroll
  for (int r = 0; r < 8; ++r) {
    float mx = -3.0e38f;
#pragma unroll
    for (int j = 0; j < 8; ++j) mx = fmaxf(mx, s_acc[j][r]);
    for (int off = 1; off < 16; off <<= 1) mx = fmaxf(mx, __shfl_xor(mx, off, 32));
    float sum = 0.f;
#pragma unroll
    for (int j = 0; j < 8; ++j) {
      float e = __expf(s_acc[j][r] - mx);
      s_acc[j][r] = e;
      sum += e;
    }
    for (int off = 1; off < 16; off <<= 1) sum += __shfl_xor(sum, off, 32);
    const float inv = 1.0f / sum;
#pragma unroll
    for (int j = 0; j < 8; ++j) s_acc[j][r] *= inv;
  }

  // spill P (D-frag layout) to wave-private LDS row-major
  _Float16* Pw = &P[wave][0];
#pragma unroll
  for (int j = 0; j < 8; ++j)
#pragma unroll
    for (int r = 0; r < 8; ++r)
      Pw[(lgrp * 8 + r) * T + j * 16 + l16] = (_Float16)s_acc[j][r];

  // Out = P·V : N = 32 (2 tiles), K = 128 (4 WMMA steps)
#pragma unroll
  for (int c = 0; c < 2; ++c) {
    v16h pa[4], vb[4];
#pragma unroll
    for (int kk = 0; kk < 4; ++kk) {
      pa[kk] = load_a_f16(Pw + kk * 32, T, l16, lgrp);
      vb[kk] = load_b_f16(Vt + (size_t)c * 16 * T + kk * 32, T, l16, lgrp);
    }
    v8f oacc = zero8();
#pragma unroll
    for (int kk = 0; kk < 4; ++kk) oacc = wmma(pa[kk], vb[kk], oacc);
#pragma unroll
    for (int r = 0; r < 8; ++r) {
      const int t = wave * 16 + lgrp * 8 + r;
      const int d = c * 16 + l16;
      const size_t off = ((((size_t)b * T + t) * H + h) * O + o) * HD + d;
      tmp[off] = (_Float16)oacc[r];
    }
  }
}

// --------------------------------------------------- spatial attention (OxO)
// grid = B*T*H = 8192; block = 128 (4 waves). Wave w owns S row-tile w (16 o's).
__global__ void __launch_bounds__(128)
spatial_kernel(const _Float16* qs, const _Float16* ks, const _Float16* tmp,
               _Float16* spat) {
  const int idx = blockIdx.x;
  const int h = idx & (H - 1), bt2 = idx >> 3, t = bt2 & (T - 1), b = bt2 >> 7;
  const size_t base = (((size_t)b * T + t) * H + h) * (size_t)(O * HD);
  const _Float16* Q = qs + base;   // (o, d) row-major, ld = 32
  const _Float16* K = ks + base;
  const _Float16* X = tmp + base;  // temporal output, same layout

  __shared__ alignas(16) _Float16 Xt[HD * O];      // transposed X: [d][o]
  __shared__ alignas(16) _Float16 P[4][16 * O];    // per-wave probs (16 x 64)

  const int tid = threadIdx.x, wave = tid >> 5, lane = tid & 31;
  const int lgrp = lane >> 4, l16 = lane & 15;

  { // stage X transposed
    const int s = tid >> 1, d0 = (tid & 1) * 16;
    v8h x0 = *(const v8h*)(X + s * HD + d0);
    v8h x1 = *(const v8h*)(X + s * HD + d0 + 8);
#pragma unroll
    for (int i = 0; i < 8; ++i) {
      Xt[(d0 + i) * O + s]     = x0[i];
      Xt[(d0 + 8 + i) * O + s] = x1[i];
    }
  }
  __syncthreads();

  const v16h qa = load_a_f16(Q + (size_t)wave * 16 * HD, HD, l16, lgrp);
  v16h kf[4];
#pragma unroll
  for (int j = 0; j < 4; ++j)
    kf[j] = load_b_f16(K + (size_t)j * 16 * HD, HD, l16, lgrp);
  v8f s_acc[4];
#pragma unroll
  for (int j = 0; j < 4; ++j) {
    s_acc[j] = zero8();
    s_acc[j] = wmma(qa, kf[j], s_acc[j]);
  }

#pragma unroll
  for (int r = 0; r < 8; ++r) {
    float mx = -3.0e38f;
#pragma unroll
    for (int j = 0; j < 4; ++j) mx = fmaxf(mx, s_acc[j][r]);
    for (int off = 1; off < 16; off <<= 1) mx = fmaxf(mx, __shfl_xor(mx, off, 32));
    float sum = 0.f;
#pragma unroll
    for (int j = 0; j < 4; ++j) {
      float e = __expf(s_acc[j][r] - mx);
      s_acc[j][r] = e;
      sum += e;
    }
    for (int off = 1; off < 16; off <<= 1) sum += __shfl_xor(sum, off, 32);
    const float inv = 1.0f / sum;
#pragma unroll
    for (int j = 0; j < 4; ++j) s_acc[j][r] *= inv;
  }

  _Float16* Pw = &P[wave][0];
#pragma unroll
  for (int j = 0; j < 4; ++j)
#pragma unroll
    for (int r = 0; r < 8; ++r)
      Pw[(lgrp * 8 + r) * O + j * 16 + l16] = (_Float16)s_acc[j][r];

  // Out = P·X : N = 32 (2 tiles), K = 64 (2 WMMA steps); merge heads on store
#pragma unroll
  for (int c = 0; c < 2; ++c) {
    v16h pa[2], vb[2];
#pragma unroll
    for (int kk = 0; kk < 2; ++kk) {
      pa[kk] = load_a_f16(Pw + kk * 32, O, l16, lgrp);
      vb[kk] = load_b_f16(Xt + (size_t)c * 16 * O + kk * 32, O, l16, lgrp);
    }
    v8f oacc = zero8();
#pragma unroll
    for (int kk = 0; kk < 2; ++kk) oacc = wmma(pa[kk], vb[kk], oacc);
#pragma unroll
    for (int r = 0; r < 8; ++r) {
      const int orow = wave * 16 + lgrp * 8 + r;
      const int d = c * 16 + l16;
      const size_t off = (((size_t)b * T + t) * O + orow) * D + h * HD + d;
      spat[off] = (_Float16)oacc[r];
    }
  }
}

// ----------------------------------------------------------- output proj GEMM
// grid = M_TOTAL/64; block = 512 (16 waves), 4 col-tiles per wave.
__global__ void __launch_bounds__(512)
outproj_kernel(const _Float16* spat, const _Float16* wo_h, const float* bo,
               float* out) {
  const int tid = threadIdx.x, wave = tid >> 5, lane = tid & 31;
  const int lgrp = lane >> 4, l16 = lane & 15;
  const int m0  = blockIdx.x * 64 + (wave >> 2) * 16;
  const int ct0 = (wave & 3) * 4;

  const _Float16* arow = spat + (size_t)(m0 + l16) * D + lgrp * 8;
  const _Float16* wrow = wo_h + ((size_t)(ct0 * 16 + l16)) * D + lgrp * 8;

  v8f acc[4];
#pragma unroll
  for (int j = 0; j < 4; ++j) acc[j] = zero8();

#pragma unroll
  for (int kb = 0; kb < D; kb += 32) {
    if (kb + 32 < D) __builtin_prefetch(arow + kb + 32, 0, 3);
    v8h alo = *(const v8h*)(arow + kb);
    v8h ahi = *(const v8h*)(arow + kb + 16);
    v16h afrag = cat8(alo, ahi);
    v16h bf[4];
#pragma unroll
    for (int j = 0; j < 4; ++j)
      bf[j] = load_b_f16(wrow + (size_t)j * 16 * D + kb, 0, 0, 0);
#pragma unroll
    for (int j = 0; j < 4; ++j) acc[j] = wmma(afrag, bf[j], acc[j]);
  }
#pragma unroll
  for (int j = 0; j < 4; ++j) {
    const int n = (ct0 + j) * 16 + l16;
    const float bv = bo[n];
#pragma unroll
    for (int r = 0; r < 8; ++r) {
      const int m = m0 + lgrp * 8 + r;
      out[(size_t)m * D + n] = acc[j][r] + bv;
    }
  }
}

} // namespace stta

extern "C" void kernel_launch(void* const* d_in, const int* in_sizes, int n_in,
                              void* d_out, int out_size, void* d_ws, size_t ws_size,
                              hipStream_t stream) {
  using namespace stta;
  (void)in_sizes; (void)n_in; (void)out_size;

  const float* q  = (const float*)d_in[0];
  const float* k  = (const float*)d_in[1];
  const float* v  = (const float*)d_in[2];
  const float* Wt = (const float*)d_in[3];
  const float* bt = (const float*)d_in[4];
  const float* Ws = (const float*)d_in[5];
  const float* bs = (const float*)d_in[6];
  const float* Wo = (const float*)d_in[7];
  const float* bo = (const float*)d_in[8];
  float* out = (float*)d_out;

  // workspace carve-up (all f16 intermediates)
  char* ws = (char*)d_ws;
  size_t off = 0;
  auto carve = [&](size_t bytes) -> char* {
    char* p = ws + off;
    off += (bytes + 255) & ~(size_t)255;
    return p;
  };
  const size_t NE = (size_t)Bn * T * O * D;  // 16,777,216 elems per tensor
  _Float16* wt_h = (_Float16*)carve(3 * D * D * sizeof(_Float16));
  _Float16* ws_h = (_Float16*)carve(2 * D * D * sizeof(_Float16));
  _Float16* wo_h = (_Float16*)carve(1 * D * D * sizeof(_Float16));
  _Float16* qt   = (_Float16*)carve(NE * sizeof(_Float16));
  _Float16* kt   = (_Float16*)carve(NE * sizeof(_Float16));
  _Float16* vt   = (_Float16*)carve(NE * sizeof(_Float16));
  _Float16* qs   = (_Float16*)carve(NE * sizeof(_Float16));
  _Float16* ks   = (_Float16*)carve(NE * sizeof(_Float16));
  _Float16* tmp  = (_Float16*)carve(NE * sizeof(_Float16));
  _Float16* spat = (_Float16*)carve(NE * sizeof(_Float16));
  if (off > ws_size) return;  // workspace too small: bail (deterministic no-op)

  // 1) pack weights to f16
  wconv_kernel<<<dim3(6 * D * D / 256), dim3(256), 0, stream>>>(
      Wt, Ws, Wo, wt_h, ws_h, wo_h);
  // 2) fused projections (5 chunks)
  proj_kernel<<<dim3(M_TOTAL / 64, 5), dim3(512), 0, stream>>>(
      q, k, v, wt_h, ws_h, bt, bs, qt, kt, vt, qs, ks);
  // 3) temporal attention
  temporal_kernel<<<dim3(Bn * O * H), dim3(256), 0, stream>>>(qt, kt, vt, tmp);
  // 4) spatial attention
  spatial_kernel<<<dim3(Bn * T * H), dim3(128), 0, stream>>>(qs, ks, tmp, spat);
  // 5) output projection (f32 result)
  outproj_kernel<<<dim3(M_TOTAL / 64), dim3(512), 0, stream>>>(spat, wo_h, bo, out);
}
